// BertNerHF_10359461118300
// MI455X (gfx1250) — matrix operand order repaired and checked by reading, not compiled, not run on hardware
//
#include <hip/hip_runtime.h>
#include <math.h>

// ---------------------------------------------------------------------------
// BERT-NER forward for MI455X (gfx1250, wave32, WMMA).
// Dense GEMMs use v_wmma_f32_16x16x32_f16 with f32 accumulation.
// A and B operands are pre-swizzled in global memory into the exact WMMA
// per-lane fragment layout so staging and fragment loads are all b128.
// Tile staging uses CDNA5 async global->LDS DMA (ASYNCcnt-tracked).
// ---------------------------------------------------------------------------

typedef __attribute__((ext_vector_type(16))) _Float16 v16h;
typedef __attribute__((ext_vector_type(8)))  float    v8f;
typedef __attribute__((ext_vector_type(4)))  int      v4i;

#if defined(__gfx1250__) && \
    __has_builtin(__builtin_amdgcn_global_load_async_to_lds_b128)
#define USE_ASYNC_LDS 1
typedef __attribute__((address_space(1))) v4i g4i;   // global v4i
typedef __attribute__((address_space(3))) v4i l4i;   // LDS v4i
#else
#define USE_ASYNC_LDS 0
#endif

constexpr int Bc  = 4;
constexpr int Sc  = 512;
constexpr int Dc  = 768;
constexpr int Lc  = 4;
constexpr int Hc  = 12;
constexpr int DHc = 64;      // Dc / Hc
constexpr int DFc = 3072;    // 4 * Dc
constexpr int NLc = 9;
constexpr int Mc  = Bc * Sc; // 2048 token rows
constexpr int MTc = Mc / 16; // 128 m-subtiles

// ---------------------------------------------------------------------------
// WMMA fragment swizzles (16x16x32 f16, wave32; CDNA5 ISA 7.12.2).
// A (16x32): lanes 0-15 hold row m%16, K {0..7,16..23}; lanes 16-31 K {8..15,
// 24..31}. Element e<8 -> K=(kh+e), e>=8 -> K=16+kh+(e-8).
// B (32x16): lanes 0-15 hold col n%16 with K {0..7,16..23}; lanes 16-31 K {8..15,24..31}.
// Layout: [k/32][tile][lane][e] -> 512 halfs per (k-tile, subtile), so one
// block tile (8 subtiles x 32 k) is a contiguous 8 KB chunk.
// ---------------------------------------------------------------------------
__device__ __forceinline__ size_t a_swz(int m, int k) {
  int kk = k & 31;
  int lane = (m & 15) + (((kk >> 3) & 1) << 4);
  int e = (kk & 7) + ((kk >> 4) << 3);
  return (((size_t)(k >> 5) * MTc + (m >> 4)) << 9) + (lane << 4) + e;
}
__device__ __forceinline__ size_t b_swz(int k, int n, int N) {
  int kk = k & 31;
  int lane = (n & 15) + (((kk >> 3) & 1) << 4);
  int e = (kk & 7) + ((kk >> 4) << 3);
  return (((size_t)(k >> 5) * (N >> 4) + (n >> 4)) << 9) + (lane << 4) + e;
}

// ---------------------------------------------------------------------------
// Block-wide sum reduction (256 threads). Caller provides 256-float LDS.
// ---------------------------------------------------------------------------
__device__ __forceinline__ float block_sum(float v, float* red) {
  int tid = threadIdx.x;
  __syncthreads();                 // protect red[] reuse across calls
  red[tid] = v;
  __syncthreads();
  #pragma unroll
  for (int off = 128; off > 0; off >>= 1) {
    if (tid < off) red[tid] += red[tid + off];
    __syncthreads();
  }
  return red[0];
}

// ---------------------------------------------------------------------------
// Weight f32 -> f16 conversion, scattering into B-fragment layout.
// ---------------------------------------------------------------------------
__global__ void cvt_bswz_kernel(const float* __restrict__ in,
                                _Float16* __restrict__ out, int N, int n) {
  int i = blockIdx.x * 256 + threadIdx.x;
  if (i < n) {
    int k = i / N, c = i % N;
    out[b_swz(k, c, N)] = (_Float16)in[i];
  }
}

// ---------------------------------------------------------------------------
// Embedding gather + LayerNorm. One block per token, 256 threads, 3 dims each.
// Writes fp32 residual stream + f16 A-fragment-swizzled copy.
// ---------------------------------------------------------------------------
__global__ void embed_ln_kernel(const int* __restrict__ ids,
                                const int* __restrict__ tids,
                                const float* __restrict__ wemb,
                                const float* __restrict__ pemb,
                                const float* __restrict__ temb,
                                const float* __restrict__ g,
                                const float* __restrict__ beta,
                                float* __restrict__ xf,
                                _Float16* __restrict__ xh) {
  int tok = blockIdx.x, tid = threadIdx.x;
  int s = tok % Sc;
  int w = ids[tok], t = tids[tok];
  __shared__ float red[256];
  float val[3];
  float loc = 0.f;
  #pragma unroll
  for (int i = 0; i < 3; ++i) {
    int d = tid + i * 256;
    val[i] = wemb[(size_t)w * Dc + d] + pemb[(size_t)s * Dc + d] +
             temb[(size_t)t * Dc + d];
    loc += val[i];
  }
  float mean = block_sum(loc, red) * (1.f / Dc);
  float lv = 0.f;
  #pragma unroll
  for (int i = 0; i < 3; ++i) { float dv = val[i] - mean; lv += dv * dv; }
  float rstd = rsqrtf(block_sum(lv, red) * (1.f / Dc) + 1e-12f);
  #pragma unroll
  for (int i = 0; i < 3; ++i) {
    int d = tid + i * 256;
    float o = (val[i] - mean) * rstd * g[d] + beta[d];
    xf[(size_t)tok * Dc + d] = o;
    xh[a_swz(tok, d)] = (_Float16)o;
  }
}

// ---------------------------------------------------------------------------
// Residual add + LayerNorm (in-place on x). delta already contains GEMM bias.
// ---------------------------------------------------------------------------
__global__ void residual_ln_kernel(const float* __restrict__ xin,
                                   const float* __restrict__ delta,
                                   const float* __restrict__ g,
                                   const float* __restrict__ beta,
                                   float* __restrict__ xf,
                                   _Float16* __restrict__ xh) {
  int tok = blockIdx.x, tid = threadIdx.x;
  __shared__ float red[256];
  float val[3];
  float loc = 0.f;
  #pragma unroll
  for (int i = 0; i < 3; ++i) {
    int d = tid + i * 256;
    val[i] = xin[(size_t)tok * Dc + d] + delta[(size_t)tok * Dc + d];
    loc += val[i];
  }
  float mean = block_sum(loc, red) * (1.f / Dc);
  float lv = 0.f;
  #pragma unroll
  for (int i = 0; i < 3; ++i) { float dv = val[i] - mean; lv += dv * dv; }
  float rstd = rsqrtf(block_sum(lv, red) * (1.f / Dc) + 1e-12f);
  #pragma unroll
  for (int i = 0; i < 3; ++i) {
    int d = tid + i * 256;
    float o = (val[i] - mean) * rstd * g[d] + beta[d];
    xf[(size_t)tok * Dc + d] = o;
    xh[a_swz(tok, d)] = (_Float16)o;
  }
}

// ---------------------------------------------------------------------------
// WMMA GEMM: C[M,N] = A[M,K] * B[K,N] + bias[N]; A/B pre-swizzled f16.
// Block tile 128x128, K-step 32, double-buffered LDS (4 x 8 KB).
// 8 wave32 waves = 4 M-strips x 2 N-strips; wave tile 32x64 = 2x4 WMMAs.
// Staging via async global->LDS DMA (ASYNCcnt); falls back to batched
// b128 load/store copies if the builtin is unavailable.
// ---------------------------------------------------------------------------
__global__ __launch_bounds__(256) void gemm_wmma_kernel(
    const _Float16* __restrict__ Asw, const _Float16* __restrict__ Bsw,
    const float* __restrict__ bias, float* __restrict__ C, int N, int K) {
  __shared__ _Float16 As[2][4096];
  __shared__ _Float16 Bs[2][4096];
  int tid = threadIdx.x;
  int lane = tid & 31, wave = tid >> 5;
  int wm = wave & 3;   // 0..3 -> 32-row strip
  int wn = wave >> 2;  // 0..1 -> 64-col strip
  int bmt = blockIdx.x * 8;           // base m-subtile
  int bnt = blockIdx.y * 8;           // base n-subtile
  int numMt = (int)gridDim.x * 8;     // M/16
  int numNt = N >> 4;
  int nk = K >> 5;

  v8f acc[2][4];
  #pragma unroll
  for (int i = 0; i < 2; ++i)
    #pragma unroll
    for (int j = 0; j < 4; ++j)
      acc[i][j] = (v8f){0.f, 0.f, 0.f, 0.f, 0.f, 0.f, 0.f, 0.f};

  // Stage one 8 KB A-chunk + one 8 KB B-chunk for k-tile kt into buffer buf.
  auto stage = [&](int kt, int buf) {
    const _Float16* ag = Asw + (((size_t)kt * numMt + bmt) << 9);
    const _Float16* bg = Bsw + (((size_t)kt * numNt + bnt) << 9);
#if USE_ASYNC_LDS
    #pragma unroll
    for (int i = 0; i < 2; ++i) {
      int o = (tid + i * 256) * 8;    // 8 halfs = 16 B per transfer
      __builtin_amdgcn_global_load_async_to_lds_b128(
          (g4i*)(ag + o), (l4i*)&As[buf][o], 0, 0);
      __builtin_amdgcn_global_load_async_to_lds_b128(
          (g4i*)(bg + o), (l4i*)&Bs[buf][o], 0, 0);
    }
#else
    float4 va[2], vb[2];
    #pragma unroll
    for (int i = 0; i < 2; ++i) {
      int o = (tid + i * 256) * 8;
      va[i] = *(const float4*)&ag[o];
      vb[i] = *(const float4*)&bg[o];
    }
    #pragma unroll
    for (int i = 0; i < 2; ++i) {
      int o = (tid + i * 256) * 8;
      *(float4*)&As[buf][o] = va[i];
      *(float4*)&Bs[buf][o] = vb[i];
    }
#endif
  };
  // Wave-local drain of outstanding async DMA before the block barrier.
  auto drain = [&]() {
#if USE_ASYNC_LDS
    asm volatile("s_wait_asynccnt 0x0" ::: "memory");
#endif
  };

  stage(0, 0);
  drain();
  __syncthreads();
  for (int kt = 0; kt < nk; ++kt) {
    int cur = kt & 1;
    if (kt + 1 < nk) stage(kt + 1, cur ^ 1);   // DMA overlaps the WMMAs

    v16h afr[2], bfr[4];
    #pragma unroll
    for (int sn = 0; sn < 4; ++sn)
      bfr[sn] = *(const v16h*)&Bs[cur][(((wn << 2) + sn) * 32 + lane) << 4];
    #pragma unroll
    for (int sm = 0; sm < 2; ++sm)
      afr[sm] = *(const v16h*)&As[cur][(((wm << 1) + sm) * 32 + lane) << 4];

    #pragma unroll
    for (int sm = 0; sm < 2; ++sm)
      #pragma unroll
      for (int sn = 0; sn < 4; ++sn)
        acc[sm][sn] = __builtin_amdgcn_wmma_f32_16x16x32_f16(
            false, afr[sm], false, bfr[sn], (short)0, acc[sm][sn], false,
            false);
    drain();
    __syncthreads();
  }

  // C/D layout: VGPR r -> row r + 8*(lane>>4), col = lane&15.
  #pragma unroll
  for (int sm = 0; sm < 2; ++sm)
    #pragma unroll
    for (int sn = 0; sn < 4; ++sn) {
      int ccol = (bnt << 4) + wn * 64 + sn * 16 + (lane & 15);
      float bv = bias[ccol];
      int rbase = (bmt << 4) + wm * 32 + sm * 16 + ((lane >> 4) << 3);
      #pragma unroll
      for (int r = 0; r < 8; ++r)
        C[(size_t)(rbase + r) * N + ccol] = acc[sm][sn][r] + bv;
    }
}

// ---------------------------------------------------------------------------
// Attention: one block per (b, h, 64-row q-tile). Scores + masked softmax +
// context. Writes context as f16 in A-fragment layout (input to Wo GEMM).
// ---------------------------------------------------------------------------
__global__ void attn_kernel(const float* __restrict__ q,
                            const float* __restrict__ k,
                            const float* __restrict__ v,
                            const int* __restrict__ mask,
                            _Float16* __restrict__ ctx) {
  constexpr int QT = 64;
  int bh = blockIdx.x / (Sc / QT);
  int qt = blockIdx.x % (Sc / QT);
  int b = bh / Hc, h = bh % Hc;
  int tid = threadIdx.x;
  __shared__ float sc[Sc];
  __shared__ float qv[DHc];
  __shared__ float red[256];
  __shared__ float bcast[2];
  const float scale = 0.125f;  // 1/sqrt(64)

  for (int qi = qt * QT; qi < qt * QT + QT; ++qi) {
    if (tid < DHc) qv[tid] = q[(size_t)(b * Sc + qi) * Dc + h * DHc + tid];
    __syncthreads();
    for (int j = tid; j < Sc; j += 256) {
      const float* kp = &k[(size_t)(b * Sc + j) * Dc + h * DHc];
      float d = 0.f;
      #pragma unroll 8
      for (int e = 0; e < DHc; ++e) d += qv[e] * kp[e];
      sc[j] = d * scale + (mask[b * Sc + j] ? 0.f : -1e9f);
    }
    __syncthreads();
    float m = -3.4e38f;
    for (int j = tid; j < Sc; j += 256) m = fmaxf(m, sc[j]);
    red[tid] = m;
    __syncthreads();
    #pragma unroll
    for (int off = 128; off > 0; off >>= 1) {
      if (tid < off) red[tid] = fmaxf(red[tid], red[tid + off]);
      __syncthreads();
    }
    if (tid == 0) bcast[0] = red[0];
    __syncthreads();
    float mx = bcast[0];
    float ssum = 0.f;
    for (int j = tid; j < Sc; j += 256) {
      float e = expf(sc[j] - mx);
      sc[j] = e;
      ssum += e;
    }
    red[tid] = ssum;
    __syncthreads();
    #pragma unroll
    for (int off = 128; off > 0; off >>= 1) {
      if (tid < off) red[tid] += red[tid + off];
      __syncthreads();
    }
    if (tid == 0) bcast[1] = 1.f / red[0];
    __syncthreads();
    float inv = bcast[1];
    int dd = tid & 63, ch = tid >> 6;
    float acc = 0.f;
    for (int j = ch * 128; j < ch * 128 + 128; ++j)
      acc += sc[j] * v[(size_t)(b * Sc + j) * Dc + h * DHc + dd];
    red[tid] = acc;
    __syncthreads();
    if (tid < DHc) {
      float r =
          (red[tid] + red[tid + 64] + red[tid + 128] + red[tid + 192]) * inv;
      ctx[a_swz(b * Sc + qi, h * DHc + tid)] = (_Float16)r;
    }
    __syncthreads();
  }
}

// ---------------------------------------------------------------------------
// Exact GELU (erf-based), f32 -> f16 A-fragment layout (input to W2 GEMM).
// ---------------------------------------------------------------------------
__global__ void gelu_kernel(const float* __restrict__ in,
                            _Float16* __restrict__ out, int n) {
  int i = blockIdx.x * 256 + threadIdx.x;
  if (i < n) {
    float x = in[i];
    float gv = 0.5f * x * (1.f + erff(x * 0.70710678118654752f));
    out[a_swz(i / DFc, i % DFc)] = (_Float16)gv;
  }
}

// ---------------------------------------------------------------------------
// Stable valid-token compaction order per batch.
// ---------------------------------------------------------------------------
__global__ void compact_kernel(const int* __restrict__ vm,
                               int* __restrict__ order,
                               int* __restrict__ nvalid) {
  int b = threadIdx.x;
  if (b >= Bc) return;
  int cnt = 0;
  for (int s = 0; s < Sc; ++s)
    if (vm[b * Sc + s]) order[b * Sc + cnt++] = s;
  nvalid[b] = cnt;
}

// ---------------------------------------------------------------------------
// Classifier + softmax over 9 labels. Row j of batch b reads gathered source
// row if j < n_valid[b], else the zero row (logits = bias).
// ---------------------------------------------------------------------------
__global__ void classify_kernel(const float* __restrict__ x,
                                const int* __restrict__ order,
                                const int* __restrict__ nvalid,
                                const float* __restrict__ W,
                                const float* __restrict__ bias,
                                float* __restrict__ out) {
  int tok = blockIdx.x, tid = threadIdx.x;
  int b = tok / Sc, j = tok % Sc;
  __shared__ float red[256];
  __shared__ float logits[NLc];
  float part[NLc];
  #pragma unroll
  for (int c = 0; c < NLc; ++c) part[c] = 0.f;
  bool valid = j < nvalid[b];  // uniform across the block
  if (valid) {
    int src = order[(size_t)b * Sc + j];
    const float* xr = &x[(size_t)(b * Sc + src) * Dc];
    for (int d = tid; d < Dc; d += 256) {
      float xv = xr[d];
      #pragma unroll
      for (int c = 0; c < NLc; ++c) part[c] += xv * W[(size_t)d * NLc + c];
    }
  }
  for (int c = 0; c < NLc; ++c) {
    __syncthreads();
    red[tid] = part[c];
    __syncthreads();
    #pragma unroll
    for (int off = 128; off > 0; off >>= 1) {
      if (tid < off) red[tid] += red[tid + off];
      __syncthreads();
    }
    if (tid == 0) logits[c] = red[0] + bias[c];
  }
  __syncthreads();
  if (tid == 0) {
    float mx = -3.4e38f;
    #pragma unroll
    for (int c = 0; c < NLc; ++c) mx = fmaxf(mx, logits[c]);
    float e[NLc];
    float s = 0.f;
    #pragma unroll
    for (int c = 0; c < NLc; ++c) { e[c] = expf(logits[c] - mx); s += e[c]; }
    float inv = 1.f / s;
    #pragma unroll
    for (int c = 0; c < NLc; ++c) out[(size_t)tok * NLc + c] = e[c] * inv;
  }
}

// ---------------------------------------------------------------------------
// Host-side orchestration. All launches on `stream`; scratch lives in d_ws.
// ---------------------------------------------------------------------------
extern "C" void kernel_launch(void* const* d_in, const int* in_sizes, int n_in,
                              void* d_out, int out_size, void* d_ws,
                              size_t ws_size, hipStream_t stream) {
  const int*   ids   = (const int*)d_in[0];
  const int*   imask = (const int*)d_in[1];
  const int*   tids  = (const int*)d_in[2];
  const int*   vmask = (const int*)d_in[3];
  const float* wemb  = (const float*)d_in[4];
  const float* pemb  = (const float*)d_in[5];
  const float* temb  = (const float*)d_in[6];
  const float* emb_g = (const float*)d_in[7];
  const float* emb_b = (const float*)d_in[8];
  const float* Wq    = (const float*)d_in[9];
  const float* bq    = (const float*)d_in[10];
  const float* Wk    = (const float*)d_in[11];
  const float* bk    = (const float*)d_in[12];
  const float* Wv    = (const float*)d_in[13];
  const float* bv    = (const float*)d_in[14];
  const float* Wo    = (const float*)d_in[15];
  const float* bo    = (const float*)d_in[16];
  const float* ln1g  = (const float*)d_in[17];
  const float* ln1b  = (const float*)d_in[18];
  const float* W1    = (const float*)d_in[19];
  const float* b1    = (const float*)d_in[20];
  const float* W2    = (const float*)d_in[21];
  const float* b2    = (const float*)d_in[22];
  const float* ln2g  = (const float*)d_in[23];
  const float* ln2b  = (const float*)d_in[24];
  const float* clsW  = (const float*)d_in[25];
  const float* clsb  = (const float*)d_in[26];
  float* out = (float*)d_out;

  // Workspace carve-out (~75 MB), 256-byte aligned slices.
  char* p = (char*)d_ws;
  auto alloc = [&](size_t bytes) -> void* {
    void* r = (void*)p;
    p += (bytes + 255) & ~(size_t)255;
    return r;
  };
  float*    xf     = (float*)alloc((size_t)Mc * Dc * 4);
  _Float16* xh     = (_Float16*)alloc((size_t)Mc * Dc * 2);   // A-swizzled
  float*    qb     = (float*)alloc((size_t)Mc * Dc * 4);
  float*    kb     = (float*)alloc((size_t)Mc * Dc * 4);
  float*    vbuf   = (float*)alloc((size_t)Mc * Dc * 4);
  _Float16* ctxh   = (_Float16*)alloc((size_t)Mc * Dc * 2);   // A-swizzled
  float*    tmp    = (float*)alloc((size_t)Mc * DFc * 4);     // shared scratch
  _Float16* hh     = (_Float16*)alloc((size_t)Mc * DFc * 2);  // A-swizzled
  _Float16* wbuf   = (_Float16*)alloc((size_t)Dc * DFc * 2);  // B-swizzled
  int*      order  = (int*)alloc((size_t)Bc * Sc * 4);
  int*      nvalid = (int*)alloc((size_t)Bc * 4);

  dim3 blk(256);
  auto cvt = [&](const float* src, _Float16* dst, int N, int n) {
    cvt_bswz_kernel<<<(n + 255) / 256, blk, 0, stream>>>(src, dst, N, n);
  };

  embed_ln_kernel<<<Mc, blk, 0, stream>>>(ids, tids, wemb, pemb, temb, emb_g,
                                          emb_b, xf, xh);

  dim3 gDD(Mc / 128, Dc / 128);    // 16 x 6
  dim3 gDF(Mc / 128, DFc / 128);   // 16 x 24

  for (int l = 0; l < Lc; ++l) {
    size_t wDD = (size_t)l * Dc * Dc;
    size_t wDF = (size_t)l * Dc * DFc;

    cvt(Wq + wDD, wbuf, Dc, Dc * Dc);
    gemm_wmma_kernel<<<gDD, blk, 0, stream>>>(xh, wbuf, bq + l * Dc, qb, Dc, Dc);
    cvt(Wk + wDD, wbuf, Dc, Dc * Dc);
    gemm_wmma_kernel<<<gDD, blk, 0, stream>>>(xh, wbuf, bk + l * Dc, kb, Dc, Dc);
    cvt(Wv + wDD, wbuf, Dc, Dc * Dc);
    gemm_wmma_kernel<<<gDD, blk, 0, stream>>>(xh, wbuf, bv + l * Dc, vbuf, Dc,
                                              Dc);

    attn_kernel<<<Bc * Hc * (Sc / 64), blk, 0, stream>>>(qb, kb, vbuf, imask,
                                                         ctxh);

    cvt(Wo + wDD, wbuf, Dc, Dc * Dc);
    gemm_wmma_kernel<<<gDD, blk, 0, stream>>>(ctxh, wbuf, bo + l * Dc, tmp, Dc,
                                              Dc);
    residual_ln_kernel<<<Mc, blk, 0, stream>>>(xf, tmp, ln1g + l * Dc,
                                               ln1b + l * Dc, xf, xh);

    cvt(W1 + wDF, wbuf, DFc, Dc * DFc);
    gemm_wmma_kernel<<<gDF, blk, 0, stream>>>(xh, wbuf, b1 + l * DFc, tmp, DFc,
                                              Dc);
    gelu_kernel<<<(Mc * DFc) / 256, blk, 0, stream>>>(tmp, hh, Mc * DFc);

    cvt(W2 + wDF, wbuf, Dc, DFc * Dc);
    gemm_wmma_kernel<<<gDD, blk, 0, stream>>>(hh, wbuf, b2 + l * Dc, tmp, Dc,
                                              DFc);
    residual_ln_kernel<<<Mc, blk, 0, stream>>>(xf, tmp, ln2g + l * Dc,
                                               ln2b + l * Dc, xf, xh);
  }

  compact_kernel<<<1, 32, 0, stream>>>(vmask, order, nvalid);
  classify_kernel<<<Mc, blk, 0, stream>>>(xf, order, nvalid, clsW, clsb, out);
}
